// Transformer_MoE_20761871909623
// MI455X (gfx1250) — compile-verified, matrix operand order checked by reading, not run on hardware
//
#include <hip/hip_runtime.h>

// ---------------------------------------------------------------------------
// Transformer + top-k MoE forward, MI455X (gfx1250), wave32 + WMMA f16.
// All GEMMs use v_wmma_f32_16x16x32_f16 (f32 accumulate). Only the top-2
// experts per token are computed (reference computes all 8 then zero-weights
// the rest). GEMM A-tiles are staged global->LDS with the CDNA5 async-LDS
// path (GLOBAL_LOAD_ASYNC_TO_LDS_B128, ASYNCcnt) and double-buffered so the
// copy overlaps the WMMA pipe. No float atomics: deterministic slot gather.
// Scratch usage ~205 MB (assumes ws_size is generous).
// ---------------------------------------------------------------------------

typedef _Float16 f16;
typedef __attribute__((ext_vector_type(16))) _Float16 v16h;
typedef __attribute__((ext_vector_type(8)))  float    v8f;

#define M_TOK   8192        // B*N
#define D_MOD   512
#define QKV_N   1536        // 3*H*Dh
#define N_HEADS 8
#define DHEAD   64
#define F_FFN   2048
#define N_EXP   8
#define SEQ     1024
#define BATCH   8
#define DEPTH   2
#define TOPK    2
#define NMB     136         // worst-case padded expert M-blocks (sum ceil/128)
#define SLOTS   (NMB*128)   // 17408

// ------------------------------ WMMA helpers -------------------------------

__device__ __forceinline__ v8f wmma_f16(v16h a, v16h b, v8f c) {
  return __builtin_amdgcn_wmma_f32_16x16x32_f16(false, a, false, b, (short)0, c,
                                                false, false);
}

// Load one 16x32 f16 fragment from LDS, "row-major" tile with leading dim ld
// (halves).  Matches the documented 16-bit A layout:
//   lanes 0-15  : row = lane,   K = {0..7, 16..23}
//   lanes 16-31 : row = lane-16,K = {8..15,24..31}
// VGPR j holds K pair ((j&4)<<2) + khalf + 2*(j&3).  The B operand is assumed
// to use the mirrored layout with "row" = output column (K-major per column);
// the tile base passed in is therefore the column-major (transposed) tile.
__device__ __forceinline__ v16h ld_frag(const f16* base, int ld) {
  int lane = threadIdx.x & 31;
  const f16* p = base + (lane & 15) * ld + ((lane >> 4) << 3);
  union { unsigned u[8]; v16h v; } r;
#pragma unroll
  for (int j = 0; j < 8; ++j) {
    int k = ((j & 4) << 2) + ((j & 3) << 1);
    r.u[j] = *(const unsigned*)(p + k);   // aligned b32 LDS read
  }
  return r.v;
}

// CDNA5 async global->LDS copy, 16 bytes per lane.  GVS addressing:
// mem = saddr(SGPR64) + vaddr(VGPR32) ; lds = VGPR32 byte address.
// Tracked by ASYNCcnt (not LOADcnt) -- no VGPR data, no loadcnt stalls.
__device__ __forceinline__ void async_lds_b128(void* lds_dst,
                                               const void* gbase,
                                               unsigned gbyteoff) {
  unsigned lds = (unsigned)(uintptr_t)lds_dst;  // low 32b of generic = LDS off
  asm volatile("global_load_async_to_lds_b128 %0, %1, %2"
               :
               : "v"(lds), "v"(gbyteoff), "s"(gbase)
               : "memory");
}
__device__ __forceinline__ void wait_asynccnt0() {
  asm volatile("s_wait_asynccnt 0x0" ::: "memory");
}

__device__ __forceinline__ float red_max16(float v) {
#pragma unroll
  for (int m = 1; m < 16; m <<= 1) v = fmaxf(v, __shfl_xor(v, m, 32));
  return v;
}
__device__ __forceinline__ float red_sum16(float v) {
#pragma unroll
  for (int m = 1; m < 16; m <<= 1) v += __shfl_xor(v, m, 32);
  return v;
}
__device__ __forceinline__ float red_sum32(float v) {
#pragma unroll
  for (int m = 1; m < 32; m <<= 1) v += __shfl_xor(v, m, 32);
  return v;
}

// ------------------------------ small kernels ------------------------------

__global__ __launch_bounds__(256) void k_copyf(float* dst, const float* src, int n) {
  int i = blockIdx.x * 256 + threadIdx.x;
  if (i < n) dst[i] = src[i];
}

__global__ __launch_bounds__(256) void k_cvt(f16* dst, const float* src, int n) {
  int i = blockIdx.x * 256 + threadIdx.x;
  if (i < n) dst[i] = (f16)src[i];
}

// LayerNorm + affine + f16 cast, one wave per 512-wide row.
__global__ __launch_bounds__(256) void k_ln_cast(const float* __restrict__ x,
                                                 const float* __restrict__ sc,
                                                 const float* __restrict__ bi,
                                                 f16* __restrict__ out) {
  int row  = blockIdx.x * 8 + (threadIdx.x >> 5);
  int lane = threadIdx.x & 31;
  const float* xr = x + (long)row * D_MOD;
  float v[16], sum = 0.f;
#pragma unroll
  for (int j = 0; j < 16; ++j) { v[j] = xr[lane + j * 32]; sum += v[j]; }
  sum = red_sum32(sum);
  float mu = sum * (1.f / D_MOD), var = 0.f;
#pragma unroll
  for (int j = 0; j < 16; ++j) { float d = v[j] - mu; var += d * d; }
  var = red_sum32(var) * (1.f / D_MOD);
  float rs = rsqrtf(var + 1e-5f);
  f16* orow = out + (long)row * D_MOD;
#pragma unroll
  for (int j = 0; j < 16; ++j) {
    int d = lane + j * 32;
    orow[d] = (f16)((v[j] - mu) * rs * sc[d] + bi[d]);
  }
}

// --------------------------- tiled WMMA GEMM -------------------------------
// C[M,N] = A[M,K] * B[K,N]  (f16 in, f32 acc).  Block tile 128x64x32, 8 waves
// in a 4x2 grid, each wave 32x32 (2x2 WMMA tiles).  A tile staged via async
// LDS copies, double-buffered to overlap global traffic with WMMA.
// Epilogues:
//   0: store f16                      (QKV)
//   1: f32 residual += acc + bias     (out projection, in-place on x)
//   2: store f16 gelu(acc + bias)     (expert W1)
//   3: store f32 acc + bias           (expert W2 -> Y)
#define BM 128
#define BN 64
#define BK 32
#define LDT (BK + 8)

template <int EPI>
__global__ __launch_bounds__(256)
void k_gemm(const f16* __restrict__ A, const f16* __restrict__ Bbase,
            const float* __restrict__ biasBase, void* __restrict__ Cout,
            int Nc, int K, const int* __restrict__ block_expert,
            long bExpStride, int biasStride) {
  __shared__ f16 As[2][BM][LDT];
  __shared__ f16 Bt[2][BN][LDT];   // transposed: Bt[n][k]

  int mb = blockIdx.y, nb = blockIdx.x;
  int e = 0;
  if (block_expert) { e = block_expert[mb]; if (e < 0) return; }
  const f16*   Bp   = Bbase + (long)e * bExpStride;
  const float* bias = biasBase ? biasBase + (long)e * biasStride : nullptr;

  int tid = threadIdx.x, lane = tid & 31, wave = tid >> 5;
  int wm = wave & 3, wn = wave >> 2;
  int rowBase = mb * BM;

  // stage A tile (async, 16B/lane, 2 issues per thread) into buffer `buf`
  auto stageA = [&](int buf, int k0) {
#pragma unroll
    for (int c = tid; c < (BM * BK / 8); c += 256) {
      int r = c >> 2, ko = (c & 3) * 8;
      async_lds_b128(&As[buf][r][ko], A,
                     (unsigned)(((long)(rowBase + r) * K + k0 + ko) * 2));
    }
  };
  // stage B tile transposed (VGPR path; async copies cannot transpose)
  auto stageB = [&](int buf, int k0) {
    int kr = tid >> 3, n0 = (tid & 7) * 8;
    const f16* src = Bp + (long)(k0 + kr) * Nc + nb * BN + n0;
#pragma unroll
    for (int j = 0; j < 8; ++j) Bt[buf][n0 + j][kr] = src[j];
  };

  v8f acc[2][2] = {};

  stageA(0, 0);
  stageB(0, 0);
  wait_asynccnt0();
  __syncthreads();

  for (int k0 = 0; k0 < K; k0 += BK) {
    int buf = (k0 >> 5) & 1;
    if (k0 + BK < K) {          // overlap next tile's copy with this compute
      stageA(buf ^ 1, k0 + BK);
      stageB(buf ^ 1, k0 + BK);
    }

    v16h af[2], bf[2];
    af[0] = ld_frag(&As[buf][wm * 32][0],      LDT);
    af[1] = ld_frag(&As[buf][wm * 32 + 16][0], LDT);
    bf[0] = ld_frag(&Bt[buf][wn * 32][0],      LDT);
    bf[1] = ld_frag(&Bt[buf][wn * 32 + 16][0], LDT);
#pragma unroll
    for (int i = 0; i < 2; ++i)
#pragma unroll
      for (int j = 0; j < 2; ++j) acc[i][j] = wmma_f16(af[i], bf[j], acc[i][j]);

    wait_asynccnt0();
    __syncthreads();
  }

  // epilogue: C layout  vgpr r -> row r (lanes<16) / r+8 (lanes>=16)
  int rofs = (lane >> 4) << 3;
#pragma unroll
  for (int i = 0; i < 2; ++i)
#pragma unroll
    for (int j = 0; j < 2; ++j)
#pragma unroll
      for (int r = 0; r < 8; ++r) {
        int row = rowBase + wm * 32 + i * 16 + r + rofs;
        int col = nb * BN + wn * 32 + j * 16 + (lane & 15);
        float v = acc[i][j][r];
        long idx = (long)row * Nc + col;
        if (EPI == 0) {
          ((f16*)Cout)[idx] = (f16)v;
        } else if (EPI == 1) {
          float* cf = (float*)Cout;
          cf[idx] = cf[idx] + v + bias[col];
        } else if (EPI == 2) {
          float h = v + bias[col];
          ((f16*)Cout)[idx] = (f16)(0.5f * h * (1.f + erff(h * 0.70710678f)));
        } else {
          ((float*)Cout)[idx] = v + bias[col];
        }
      }
}

// ------------------------- flash attention (WMMA) --------------------------
#define AT_BQ 128
#define AT_BK 64
#define LDA 72   // DHEAD+8 == AT_BK+8

__global__ __launch_bounds__(256) void k_attn(const f16* __restrict__ qkv,
                                              f16* __restrict__ o) {
  __shared__ f16 Qs[AT_BQ][LDA];
  __shared__ f16 Ks[AT_BK][LDA];
  __shared__ f16 Vt[DHEAD][LDA];   // transposed: Vt[dh][key]
  __shared__ f16 Ps[AT_BQ][LDA];

  int qb = blockIdx.x, h = blockIdx.y, b = blockIdx.z;
  int tid = threadIdx.x, lane = tid & 31, wave = tid >> 5;
  const long rs = QKV_N;
  const f16* qbase = qkv + ((long)b * SEQ + qb * AT_BQ) * rs + h * DHEAD;

#pragma unroll
  for (int c = tid; c < AT_BQ * DHEAD / 8; c += 256) {
    int r = c >> 3, d0 = (c & 7) * 8;
    *(uint4*)&Qs[r][d0] = *(const uint4*)(qbase + (long)r * rs + d0);
  }

  v8f Oa[4] = {};
  float mrow[8], lrow[8];
#pragma unroll
  for (int r = 0; r < 8; ++r) { mrow[r] = -__builtin_inff(); lrow[r] = 0.f; }

  for (int kt = 0; kt < SEQ; kt += AT_BK) {
    __syncthreads();
    const f16* kbase = qkv + ((long)b * SEQ + kt) * rs + D_MOD + h * DHEAD;
    const f16* vbase = kbase + D_MOD;
#pragma unroll
    for (int c = tid; c < AT_BK * DHEAD / 8; c += 256) {
      int r = c >> 3, d0 = (c & 7) * 8;
      *(uint4*)&Ks[r][d0] = *(const uint4*)(kbase + (long)r * rs + d0);
      uint4 vv = *(const uint4*)(vbase + (long)r * rs + d0);
      const f16* vh = (const f16*)&vv;
#pragma unroll
      for (int j = 0; j < 8; ++j) Vt[d0 + j][r] = vh[j];
    }
    __syncthreads();

    // S = (Q K^T) * Dh^-0.5   (16 q rows x 64 keys per wave)
    v8f S[4] = {};
#pragma unroll
    for (int kk = 0; kk < 2; ++kk) {
      v16h aq = ld_frag(&Qs[wave * 16][kk * 32], LDA);
#pragma unroll
      for (int nt = 0; nt < 4; ++nt) {
        v16h bk = ld_frag(&Ks[nt * 16][kk * 32], LDA);
        S[nt] = wmma_f16(aq, bk, S[nt]);
      }
    }

    // online softmax update, all in accumulator layout
#pragma unroll
    for (int r = 0; r < 8; ++r) {
      float mx = -__builtin_inff();
#pragma unroll
      for (int nt = 0; nt < 4; ++nt) { S[nt][r] *= 0.125f; mx = fmaxf(mx, S[nt][r]); }
      mx = red_max16(mx);
      float mnew = fmaxf(mrow[r], mx);
      float fac  = __expf(mrow[r] - mnew);
      mrow[r] = mnew;
      float ps = 0.f;
#pragma unroll
      for (int nt = 0; nt < 4; ++nt) {
        float p = __expf(S[nt][r] - mnew);
        S[nt][r] = p; ps += p;
      }
      ps = red_sum16(ps);
      lrow[r] = lrow[r] * fac + ps;
#pragma unroll
      for (int dt = 0; dt < 4; ++dt) Oa[dt][r] *= fac;
    }

    // spill P to LDS (f16) to reshape into A fragments
    int rofs = (lane >> 4) << 3;
#pragma unroll
    for (int nt = 0; nt < 4; ++nt)
#pragma unroll
      for (int r = 0; r < 8; ++r)
        Ps[wave * 16 + r + rofs][nt * 16 + (lane & 15)] = (f16)S[nt][r];
    __syncthreads();

    // O += P V
#pragma unroll
    for (int kc = 0; kc < 2; ++kc) {
      v16h ap = ld_frag(&Ps[wave * 16][kc * 32], LDA);
#pragma unroll
      for (int dt = 0; dt < 4; ++dt) {
        v16h bv = ld_frag(&Vt[dt * 16][kc * 32], LDA);
        Oa[dt] = wmma_f16(ap, bv, Oa[dt]);
      }
    }
  }

  int rofs = (lane >> 4) << 3;
#pragma unroll
  for (int dt = 0; dt < 4; ++dt)
#pragma unroll
    for (int r = 0; r < 8; ++r) {
      int row = qb * AT_BQ + wave * 16 + r + rofs;
      float val = Oa[dt][r] / lrow[r];
      o[((long)b * SEQ + row) * D_MOD + h * DHEAD + dt * 16 + (lane & 15)] = (f16)val;
    }
}

// ------------------------------- routing -----------------------------------

__global__ __launch_bounds__(256) void k_reset(int* row_token, int* counts, int* cursors) {
  int i = blockIdx.x * 256 + threadIdx.x;
  if (i < SLOTS) row_token[i] = -1;
  if (i < N_EXP) { counts[i] = 0; cursors[i] = 0; }
}

__global__ __launch_bounds__(256) void k_router(const float* __restrict__ x,
                                                const float* __restrict__ rtw,
                                                const float* __restrict__ rtb,
                                                float* __restrict__ wout,
                                                int* __restrict__ tok_e,
                                                float* __restrict__ tok_g,
                                                int* __restrict__ counts) {
  int t = blockIdx.x * 256 + threadIdx.x;
  const float* xr = x + (long)t * D_MOD;
  float lg[N_EXP];
#pragma unroll
  for (int e = 0; e < N_EXP; ++e) lg[e] = rtb[e];
  for (int d = 0; d < D_MOD; ++d) {
    float xv = xr[d];
#pragma unroll
    for (int e = 0; e < N_EXP; ++e) lg[e] += xv * rtw[d * N_EXP + e];
  }
  int i0 = 0; float v0 = lg[0];
#pragma unroll
  for (int e = 1; e < N_EXP; ++e) if (lg[e] > v0) { v0 = lg[e]; i0 = e; }
  int i1 = -1; float v1 = -__builtin_inff();
#pragma unroll
  for (int e = 0; e < N_EXP; ++e)
    if (e != i0 && lg[e] > v1) { v1 = lg[e]; i1 = e; }
  float p1 = __expf(v1 - v0), inv = 1.f / (1.f + p1);
  float g0 = inv, g1 = p1 * inv;
#pragma unroll
  for (int e = 0; e < N_EXP; ++e) wout[(long)t * N_EXP + e] = 0.f;
  wout[(long)t * N_EXP + i0] = g0;
  wout[(long)t * N_EXP + i1] = g1;
  tok_e[t * 2] = i0; tok_e[t * 2 + 1] = i1;
  tok_g[t * 2] = g0; tok_g[t * 2 + 1] = g1;
  atomicAdd(&counts[i0], 1);
  atomicAdd(&counts[i1], 1);
}

__global__ void k_offsets(const int* counts, int* offsets, int* block_expert) {
  if (threadIdx.x == 0 && blockIdx.x == 0) {
    int off = 0, mb = 0;
    for (int e = 0; e < N_EXP; ++e) {
      offsets[e] = off;
      int nb = (counts[e] + 127) >> 7;
      off += nb << 7;
      for (int j = 0; j < nb; ++j) block_expert[mb++] = e;
    }
    for (; mb < NMB; ++mb) block_expert[mb] = -1;
  }
}

__global__ __launch_bounds__(256) void k_assign(const int* tok_e, const float* tok_g,
                                                const int* offsets, int* cursors,
                                                int* row_token, int* slot_of) {
  int idx = blockIdx.x * 256 + threadIdx.x;
  if (idx >= M_TOK * TOPK) return;
  int e = tok_e[idx];
  int pos = atomicAdd(&cursors[e], 1);
  int slot = offsets[e] + pos;
  row_token[slot] = idx >> 1;
  slot_of[idx] = slot;
}

// gather x[token] -> LN -> per-expert affine -> f16, one wave per slot row
__global__ __launch_bounds__(256) void k_gather_ln2(const float* __restrict__ x,
                                                    const int* __restrict__ row_token,
                                                    const int* __restrict__ block_expert,
                                                    const float* __restrict__ ln2_s,
                                                    const float* __restrict__ ln2_b,
                                                    f16* __restrict__ Xe) {
  int slot = blockIdx.x * 8 + (threadIdx.x >> 5);
  int lane = threadIdx.x & 31;
  int t = row_token[slot];
  int e = block_expert[slot >> 7];
  f16* orow = Xe + (long)slot * D_MOD;
  if (t < 0 || e < 0) {
#pragma unroll
    for (int j = 0; j < 16; ++j) orow[lane + j * 32] = (f16)0.f;
    return;
  }
  const float* xr = x + (long)t * D_MOD;
  float v[16], sum = 0.f;
#pragma unroll
  for (int j = 0; j < 16; ++j) { v[j] = xr[lane + j * 32]; sum += v[j]; }
  sum = red_sum32(sum);
  float mu = sum * (1.f / D_MOD), var = 0.f;
#pragma unroll
  for (int j = 0; j < 16; ++j) { float d = v[j] - mu; var += d * d; }
  var = red_sum32(var) * (1.f / D_MOD);
  float rsq = rsqrtf(var + 1e-5f);
  const float* sc = ln2_s + (long)e * D_MOD;
  const float* bi = ln2_b + (long)e * D_MOD;
#pragma unroll
  for (int j = 0; j < 16; ++j) {
    int d = lane + j * 32;
    orow[d] = (f16)((v[j] - mu) * rsq * sc[d] + bi[d]);
  }
}

// x += g0*Y[slot0] + g1*Y[slot1]  (deterministic, no atomics)
__global__ __launch_bounds__(256) void k_combine(float* __restrict__ x,
                                                 const float* __restrict__ Y,
                                                 const int* __restrict__ slot_of,
                                                 const float* __restrict__ tok_g) {
  long i = (long)blockIdx.x * 256 + threadIdx.x;
  int t = (int)(i >> 9), d = (int)(i & 511);
  int s0 = slot_of[t * 2], s1 = slot_of[t * 2 + 1];
  float g0 = tok_g[t * 2], g1 = tok_g[t * 2 + 1];
  x[i] += g0 * Y[(long)s0 * D_MOD + d] + g1 * Y[(long)s1 * D_MOD + d];
}

// ------------------------------- host side ---------------------------------

extern "C" void kernel_launch(void* const* d_in, const int* in_sizes, int n_in,
                              void* d_out, int out_size, void* d_ws, size_t ws_size,
                              hipStream_t stream) {
  const float* x_in  = (const float*)d_in[0];
  const float* ln1_s = (const float*)d_in[1];
  const float* ln1_b = (const float*)d_in[2];
  const float* qkvw  = (const float*)d_in[3];
  const float* outw  = (const float*)d_in[4];
  const float* outb  = (const float*)d_in[5];
  const float* rtw   = (const float*)d_in[6];
  const float* rtb   = (const float*)d_in[7];
  const float* ln2s  = (const float*)d_in[8];
  const float* ln2b  = (const float*)d_in[9];
  const float* w1    = (const float*)d_in[10];
  const float* b1    = (const float*)d_in[11];
  const float* w2    = (const float*)d_in[12];
  const float* b2    = (const float*)d_in[13];

  float* x    = (float*)d_out;                       // residual stream [M,D]
  float* wout = (float*)d_out + (long)M_TOK * D_MOD; // [depth,M,E]

  char* ws = (char*)d_ws;
  auto alloc = [&](size_t bytes) -> char* {
    char* p = ws; ws += (bytes + 255) & ~(size_t)255; return p;
  };
  f16*   xn16    = (f16*)alloc((size_t)M_TOK * D_MOD * 2);
  f16*   qkv16   = (f16*)alloc((size_t)M_TOK * QKV_N * 2);
  f16*   o16     = (f16*)alloc((size_t)M_TOK * D_MOD * 2);
  f16*   qkvw16  = (f16*)alloc((size_t)D_MOD * QKV_N * 2);
  f16*   outw16  = (f16*)alloc((size_t)D_MOD * D_MOD * 2);
  f16*   w1_16   = (f16*)alloc((size_t)N_EXP * D_MOD * F_FFN * 2);
  f16*   w2_16   = (f16*)alloc((size_t)N_EXP * F_FFN * D_MOD * 2);
  f16*   Xe16    = (f16*)alloc((size_t)SLOTS * D_MOD * 2);
  f16*   H16     = (f16*)alloc((size_t)SLOTS * F_FFN * 2);
  float* Yf      = (float*)alloc((size_t)SLOTS * D_MOD * 4);
  int*   tok_e   = (int*)alloc((size_t)M_TOK * TOPK * 4);
  float* tok_g   = (float*)alloc((size_t)M_TOK * TOPK * 4);
  int*   slot_of = (int*)alloc((size_t)M_TOK * TOPK * 4);
  int*   row_tok = (int*)alloc((size_t)SLOTS * 4);
  int*   counts  = (int*)alloc(64 * 4);
  int*   cursors = (int*)alloc(64 * 4);
  int*   offsets = (int*)alloc(64 * 4);
  int*   blk_e   = (int*)alloc((size_t)NMB * 4);
  (void)in_sizes; (void)n_in; (void)out_size; (void)ws_size;

  k_copyf<<<M_TOK * D_MOD / 256, 256, 0, stream>>>(x, x_in, M_TOK * D_MOD);

  for (int L = 0; L < DEPTH; ++L) {
    // per-layer f16 weight conversion (hot in L2 afterwards)
    int nqkv = D_MOD * QKV_N, nout = D_MOD * D_MOD;
    int nw   = N_EXP * D_MOD * F_FFN;
    k_cvt<<<(nqkv + 255) / 256, 256, 0, stream>>>(qkvw16, qkvw + (long)L * nqkv, nqkv);
    k_cvt<<<(nout + 255) / 256, 256, 0, stream>>>(outw16, outw + (long)L * nout, nout);
    k_cvt<<<(nw + 255) / 256, 256, 0, stream>>>(w1_16, w1 + (long)L * nw, nw);
    k_cvt<<<(nw + 255) / 256, 256, 0, stream>>>(w2_16, w2 + (long)L * nw, nw);

    // LN1 + QKV + attention + out-projection (residual)
    k_ln_cast<<<M_TOK / 8, 256, 0, stream>>>(x, ln1_s + L * D_MOD,
                                             ln1_b + L * D_MOD, xn16);
    k_gemm<0><<<dim3(QKV_N / BN, M_TOK / BM), 256, 0, stream>>>(
        xn16, qkvw16, nullptr, qkv16, QKV_N, D_MOD, nullptr, 0, 0);
    k_attn<<<dim3(SEQ / AT_BQ, N_HEADS, BATCH), 256, 0, stream>>>(qkv16, o16);
    k_gemm<1><<<dim3(D_MOD / BN, M_TOK / BM), 256, 0, stream>>>(
        o16, outw16, outb + L * D_MOD, x, D_MOD, D_MOD, nullptr, 0, 0);

    // routing: top-2 + weights output + compact per-expert slot lists
    k_reset<<<SLOTS / 256, 256, 0, stream>>>(row_tok, counts, cursors);
    k_router<<<M_TOK / 256, 256, 0, stream>>>(
        x, rtw + (long)L * D_MOD * N_EXP, rtb + L * N_EXP,
        wout + (long)L * M_TOK * N_EXP, tok_e, tok_g, counts);
    k_offsets<<<1, 32, 0, stream>>>(counts, offsets, blk_e);
    k_assign<<<M_TOK * TOPK / 256, 256, 0, stream>>>(tok_e, tok_g, offsets,
                                                     cursors, row_tok, slot_of);
    k_gather_ln2<<<SLOTS / 8, 256, 0, stream>>>(
        x, row_tok, blk_e, ln2s + (long)L * N_EXP * D_MOD,
        ln2b + (long)L * N_EXP * D_MOD, Xe16);

    // expert FFN on the ~2/8 active expert rows only
    k_gemm<2><<<dim3(F_FFN / BN, NMB), 256, 0, stream>>>(
        Xe16, w1_16, b1 + (long)L * N_EXP * F_FFN, H16, F_FFN, D_MOD,
        blk_e, (long)D_MOD * F_FFN, F_FFN);
    k_gemm<3><<<dim3(D_MOD / BN, NMB), 256, 0, stream>>>(
        H16, w2_16, b2 + (long)L * N_EXP * D_MOD, Yf, D_MOD, F_FFN,
        blk_e, (long)F_FFN * D_MOD, D_MOD);
    k_combine<<<M_TOK * D_MOD / 256, 256, 0, stream>>>(x, Yf, slot_of, tok_g);
  }
}
